// ECCPagedAttentionShim_80058190397993
// MI455X (gfx1250) — matrix operand order repaired and checked by reading, not compiled, not run on hardware
//
#include <hip/hip_runtime.h>

// Problem constants (must match reference)
#define B_     8
#define Q_     16
#define S_     4096
#define H_     32
#define KVH_   8
#define D_     128
#define G_     4        // H / KVH
#define NSPLIT 4
#define CHUNK  (S_ / NSPLIT)   // 1024 keys per split
#define TILE   32              // keys per inner tile (== WMMA K for PV)

typedef __attribute__((ext_vector_type(16))) _Float16     v16h;
typedef __attribute__((ext_vector_type(8)))  float        v8f;
typedef __attribute__((ext_vector_type(4)))  float        f32x4;
typedef __attribute__((ext_vector_type(4)))  unsigned int u32x4;

union FragU { v16h h; u32x4 u[2]; };   // trivial types -> trivially constructible union

__device__ __forceinline__ v16h lds_load_v16h(const _Float16* p) {
  FragU f;
  f.u[0] = *(const u32x4*)(p);
  f.u[1] = *(const u32x4*)(p + 8);
  return f.h;
}

// DPP16 butterfly max step: combine with lane (lane ^ pattern) inside a 16-lane row.
// 0xB1 = quad_perm [1,0,3,2] (xor1), 0x4E = quad_perm [2,3,0,1] (xor2),
// 0x141 = row_half_mirror (xor7), 0x140 = row_mirror (xor15).
template <int CTRL>
__device__ __forceinline__ float dpp_max_step(float x) {
  const int y = __builtin_amdgcn_update_dpp(0, __float_as_int(x), CTRL, 0xF, 0xF, true);
  return fmaxf(x, __int_as_float(y));
}

// ---------------------------------------------------------------------------
// Pass 1: fused INT4 quant/dequant + flash decode attention, split over S.
// Grid: (B*KVH*NSPLIT) blocks of 128 threads (4 waves; wave w = GQA head g=w).
// Software-pipelined: global loads for tile t+1 are issued before the WMMA
// work of tile t, so HBM latency overlaps matrix math.
// ---------------------------------------------------------------------------
__global__ __launch_bounds__(128)
void ecc_paged_attn_partial(const float* __restrict__ q,
                            const float* __restrict__ k,
                            const float* __restrict__ v,
                            float* __restrict__ part_o,   // [64*NSPLIT][64][128]
                            float* __restrict__ part_m,   // [64*NSPLIT][64]
                            float* __restrict__ part_l)   // [64*NSPLIT][64]
{
  __shared__ __align__(16) _Float16 kq  [TILE * D_];      // quantized K (nib-8), [token][d]
  __shared__ __align__(16) _Float16 vqT [D_ * TILE];      // dequantized V (c*scale_v), [d][token]
  __shared__ __align__(16) _Float16 plds[4 * 16 * TILE];  // per-wave P scratch
  __shared__ float sck[TILE];

  const int blk   = blockIdx.x;
  const int split = blk & (NSPLIT - 1);
  const int bkvh  = blk >> 2;          // NSPLIT == 4
  const int b     = bkvh >> 3;         // KVH == 8
  const int kvh   = bkvh & 7;
  const int tid   = threadIdx.x;
  const int wave  = tid >> 5;          // GQA sub-head g
  const int lane  = tid & 31;
  const int half_ = lane >> 4;
  const int nidx  = lane & 15;
  const int h     = kvh * G_ + wave;

  // ---- Q A-fragments: 16 rows (qi = nidx) of head h, f16 --------------------
  // A 16x32 layout: lane%16 = M, element j -> K = 16*half + j
  v16h aq[4];
  {
    const float* qrow = q + (((size_t)(b * Q_ + nidx) * H_ + h) * (size_t)D_);
    #pragma unroll
    for (int kc = 0; kc < 4; ++kc) {
      const float* p0 = qrow + kc * 32 + half_ * 16;
      const float* p1 = qrow + kc * 32 + half_ * 16 + 8;
      f32x4 a0 = *(const f32x4*)(p0);
      f32x4 a1 = *(const f32x4*)(p0 + 4);
      f32x4 a2 = *(const f32x4*)(p1);
      f32x4 a3 = *(const f32x4*)(p1 + 4);
      float tmp[16];
      #pragma unroll
      for (int j = 0; j < 4; ++j) {
        tmp[j] = a0[j]; tmp[4 + j] = a1[j]; tmp[8 + j] = a2[j]; tmp[12 + j] = a3[j];
      }
      v16h a;
      #pragma unroll
      for (int j = 0; j < 16; ++j) a[j] = (_Float16)tmp[j];
      aq[kc] = a;
    }
  }

  // all-ones B fragment: row-sum of P via one extra WMMA (l bookkeeping)
  v16h ones;
  #pragma unroll
  for (int j = 0; j < 16; ++j) ones[j] = (_Float16)1.0f;

  float m_run[8], l_run[8];
  #pragma unroll
  for (int r = 0; r < 8; ++r) { m_run[r] = -3.0e38f; l_run[r] = 0.0f; }
  v8f o_acc[8];
  #pragma unroll
  for (int f = 0; f < 8; ++f) { v8f z = {}; o_acc[f] = z; }

  const int    ntiles  = CHUNK / TILE;
  const int    s_base  = split * CHUNK;
  const float  rsqD    = 0.08838834764831845f;   // 1/sqrt(128)
  const int    tok     = tid >> 2;               // token within tile (0..31)
  const int    d0      = (tid & 3) * 32;         // 32 d-elements per thread
  const size_t sstride = (size_t)KVH_ * D_;      // float stride per token
  const size_t base0   = (((size_t)(b * S_ + tok)) * KVH_ + kvh) * (size_t)D_ + d0;

  // ---- preload tile 0 into registers ----------------------------------------
  float kf[32], vf[32];
  {
    const size_t g0 = base0 + (size_t)s_base * sstride;
    #pragma unroll
    for (int i = 0; i < 8; ++i) {
      f32x4 ka = *(const f32x4*)(k + g0 + i * 4);
      f32x4 va = *(const f32x4*)(v + g0 + i * 4);
      #pragma unroll
      for (int j = 0; j < 4; ++j) { kf[i * 4 + j] = ka[j]; vf[i * 4 + j] = va[j]; }
    }
  }

  for (int t = 0; t < ntiles; ++t) {
    const int s0 = s_base + t * TILE;

    // ---- quantize tile t in registers (absmax across the 4 quad threads) ----
    float amk = 0.0f, amv = 0.0f;
    #pragma unroll
    for (int i = 0; i < 32; ++i) {
      amk = fmaxf(amk, fabsf(kf[i]));
      amv = fmaxf(amv, fabsf(vf[i]));
    }
    amk = dpp_max_step<0xB1>(amk); amk = dpp_max_step<0x4E>(amk);
    amv = dpp_max_step<0xB1>(amv); amv = dpp_max_step<0x4E>(amv);
    const float sk  = fmaxf(amk * (1.0f / 7.0f), 1e-8f);
    const float sv  = fmaxf(amv * (1.0f / 7.0f), 1e-8f);
    const float rsk = 1.0f / sk;
    const float rsv = 1.0f / sv;
    __align__(16) _Float16 hk[32];
    _Float16 hv[32];
    #pragma unroll
    for (int i = 0; i < 32; ++i) {
      hk[i] = (_Float16)fminf(fmaxf(rintf(kf[i] * rsk), -8.0f), 7.0f);
      hv[i] = (_Float16)(fminf(fmaxf(rintf(vf[i] * rsv), -8.0f), 7.0f) * sv);
    }

    __syncthreads();   // previous tile fully consumed before overwrite
    {
      u32x4*       dst = (u32x4*)&kq[tok * D_ + d0];
      const u32x4* src = (const u32x4*)hk;
      #pragma unroll
      for (int i = 0; i < 4; ++i) dst[i] = src[i];
      #pragma unroll
      for (int i = 0; i < 32; ++i) vqT[(d0 + i) * TILE + tok] = hv[i];
      if ((tid & 3) == 0) sck[tok] = sk;
    }
    __syncthreads();

    // ---- issue global loads for tile t+1 (consumed next iteration) ----------
    if (t + 1 < ntiles) {
      const size_t g1 = base0 + (size_t)(s0 + TILE) * sstride;
      #pragma unroll
      for (int i = 0; i < 8; ++i) {
        f32x4 ka = *(const f32x4*)(k + g1 + i * 4);
        f32x4 va = *(const f32x4*)(v + g1 + i * 4);
        #pragma unroll
        for (int j = 0; j < 4; ++j) { kf[i * 4 + j] = ka[j]; vf[i * 4 + j] = va[j]; }
      }
      __builtin_prefetch(k + g1 + (size_t)TILE * sstride, 0, 0);
      __builtin_prefetch(v + g1 + (size_t)TILE * sstride, 0, 0);
    }

    // ---- Q.K^T : two 16-key N-subtiles, 4 K-chunks of 32 --------------------
    v8f acc0 = {}, acc1 = {};
    #pragma unroll
    for (int kc = 0; kc < 4; ++kc) {
      // B 32x16 layout: N = lane%16 (key), element j -> K(d) = 16*half + j
      const v16h b0 = lds_load_v16h(&kq[nidx * D_        + kc * 32 + half_ * 16]);
      const v16h b1 = lds_load_v16h(&kq[(16 + nidx) * D_ + kc * 32 + half_ * 16]);
      acc0 = __builtin_amdgcn_wmma_f32_16x16x32_f16(false, aq[kc], false, b0,
                                                    (short)0, acc0, false, false);
      acc1 = __builtin_amdgcn_wmma_f32_16x16x32_f16(false, aq[kc], false, b1,
                                                    (short)0, acc1, false, false);
    }

    const float sk0 = sck[nidx]      * rsqD;
    const float sk1 = sck[16 + nidx] * rsqD;
    const bool  lastTile = (s0 + TILE) == S_;   // only tile needing causal mask

    float p0e[8], p1e[8], corr[8];
    #pragma unroll
    for (int r = 0; r < 8; ++r) {
      const int m = r + half_ * 8;              // C-layout: row = VGPR + 8*half
      float x0 = acc0[r] * sk0;
      float x1 = acc1[r] * sk1;
      if (lastTile && (nidx > m)) x1 = -1.0e30f;  // key s = S-16+nidx vs qpos S-16+m
      float rm = fmaxf(x0, x1);
      rm = dpp_max_step<0xB1>(rm);
      rm = dpp_max_step<0x4E>(rm);
      rm = dpp_max_step<0x141>(rm);
      rm = dpp_max_step<0x140>(rm);
      const float mnew = fmaxf(m_run[r], rm);
      corr[r]  = __expf(m_run[r] - mnew);
      p0e[r]   = __expf(x0 - mnew);
      p1e[r]   = __expf(x1 - mnew);
      m_run[r] = mnew;
    }
    #pragma unroll
    for (int f = 0; f < 8; ++f) {
      #pragma unroll
      for (int r = 0; r < 8; ++r) o_acc[f][r] *= corr[r];
    }

    // ---- P (pure e-values) -> per-wave LDS, reload in A-layout --------------
    _Float16* pl = &plds[wave * (16 * TILE)];
    #pragma unroll
    for (int r = 0; r < 8; ++r) {
      const int m = r + half_ * 8;
      pl[m * TILE + nidx]      = (_Float16)p0e[r];
      pl[m * TILE + 16 + nidx] = (_Float16)p1e[r];
    }
    // same-wave DS ops are in-order: store->load RAW through LDS is safe
    const _Float16* prow = &plds[wave * (16 * TILE) + nidx * TILE];
    FragU fa;
    fa.u[0] = *(const u32x4*)(prow + half_ * 16);
    fa.u[1] = *(const u32x4*)(prow + half_ * 16 + 8);

    // ---- l row-sum via one WMMA against all-ones B --------------------------
    v8f lz = {};
    const v8f lsum = __builtin_amdgcn_wmma_f32_16x16x32_f16(false, fa.h, false, ones,
                                                            (short)0, lz, false, false);

    // ---- P.V : D=128 as 8 N-chunks of 16, K = 32 keys (scale_v pre-folded) --
    #pragma unroll
    for (int dc = 0; dc < 8; ++dc) {
      const v16h bv = lds_load_v16h(&vqT[(dc * 16 + nidx) * TILE + half_ * 16]);
      o_acc[dc] = __builtin_amdgcn_wmma_f32_16x16x32_f16(false, fa.h, false, bv,
                                                         (short)0, o_acc[dc],
                                                         false, false);
    }

    #pragma unroll
    for (int r = 0; r < 8; ++r) l_run[r] = l_run[r] * corr[r] + lsum[r];
  }

  // ---- emit unnormalized partials + (m, l) stats -----------------------------
  const size_t rowbase = ((size_t)(bkvh * NSPLIT + split) * 64) + wave * 16;
  #pragma unroll
  for (int dc = 0; dc < 8; ++dc) {
    #pragma unroll
    for (int r = 0; r < 8; ++r) {
      const int m = r + half_ * 8;
      part_o[(rowbase + m) * D_ + dc * 16 + nidx] = o_acc[dc][r];
    }
  }
  if (nidx == 0) {
    #pragma unroll
    for (int r = 0; r < 8; ++r) {
      const int m = r + half_ * 8;
      part_m[rowbase + m] = m_run[r];
      part_l[rowbase + m] = l_run[r];
    }
  }
}

// ---------------------------------------------------------------------------
// Pass 2: merge NSPLIT partial softmax results (flash-decode combine).
// ---------------------------------------------------------------------------
__global__ __launch_bounds__(256)
void ecc_paged_attn_combine(const float* __restrict__ part_o,
                            const float* __restrict__ part_m,
                            const float* __restrict__ part_l,
                            float* __restrict__ out)
{
  const int total = 64 * 64 * D_;               // (B*KVH) * (G*Q) * D
  const int idx   = blockIdx.x * 256 + threadIdx.x;
  if (idx >= total) return;
  const int d    = idx & (D_ - 1);
  const int row  = (idx >> 7) & 63;             // g*16 + qi
  const int bkvh = idx >> 13;

  float mi[NSPLIT];
  float M = -3.0e38f;
  #pragma unroll
  for (int s = 0; s < NSPLIT; ++s) {
    mi[s] = part_m[((size_t)(bkvh * NSPLIT + s)) * 64 + row];
    M = fmaxf(M, mi[s]);
  }
  float num = 0.0f, den = 0.0f;
  #pragma unroll
  for (int s = 0; s < NSPLIT; ++s) {
    const float  w  = __expf(mi[s] - M);
    const size_t rb = ((size_t)(bkvh * NSPLIT + s)) * 64 + row;
    num += part_o[rb * D_ + d] * w;
    den += part_l[rb] * w;
  }
  const int g = row >> 4, qi = row & 15;
  const int b = bkvh >> 3, kvh = bkvh & 7;
  out[(((size_t)(b * Q_ + qi) * H_) + kvh * G_ + g) * (size_t)D_ + d] = num / den;
}

// ---------------------------------------------------------------------------
extern "C" void kernel_launch(void* const* d_in, const int* in_sizes, int n_in,
                              void* d_out, int out_size, void* d_ws, size_t ws_size,
                              hipStream_t stream) {
  (void)in_sizes; (void)n_in; (void)out_size; (void)ws_size;
  const float* q = (const float*)d_in[0];
  const float* k = (const float*)d_in[1];
  const float* v = (const float*)d_in[2];
  // d_in[3] (block_table): gather(bt) o scatter(bt) is identity for any
  // permutation table, and Hamming(8,4) decode(encode(x)) == x, so the cache
  // round-trip reduces to INT4 quant/dequant fused into the attention pass.

  float* part_o = (float*)d_ws;                                   // 8 MB
  float* part_m = part_o + (size_t)64 * NSPLIT * 64 * D_;         // 64 KB
  float* part_l = part_m + (size_t)64 * NSPLIT * 64;              // 64 KB

  ecc_paged_attn_partial<<<dim3(64 * NSPLIT), dim3(128), 0, stream>>>(
      q, k, v, part_o, part_m, part_l);

  const int total = 64 * 64 * D_;
  ecc_paged_attn_combine<<<dim3((total + 255) / 256), dim3(256), 0, stream>>>(
      part_o, part_m, part_l, (float*)d_out);
}
